// GNNNet_84516366451356
// MI455X (gfx1250) — compile-verified, matrix operand order checked by reading, not run on hardware
//
#include <hip/hip_runtime.h>
#include <hip/hip_bf16.h>

// ---------------- constants ----------------
#define NN 40000
#define EE 640000
#define DOUT_FINAL 10

typedef __attribute__((ext_vector_type(16))) __bf16   v16bf;
typedef __attribute__((ext_vector_type(8)))  float    v8f;
typedef __attribute__((ext_vector_type(8)))  unsigned v8u;

// round-to-nearest-even f32 -> bf16 bit pattern
__device__ __forceinline__ unsigned bf16bits(float f) {
  unsigned u = __builtin_bit_cast(unsigned, f);
  return (u + 0x7FFFu + ((u >> 16) & 1u)) >> 16;
}

// order-preserving float<->uint encoding for atomicMax-based segment max
__device__ __forceinline__ unsigned fkey(float f) {
  unsigned u = __builtin_bit_cast(unsigned, f);
  return (u & 0x80000000u) ? ~u : (u | 0x80000000u);
}
__device__ __forceinline__ float funkey(unsigned k) {
  unsigned u = (k & 0x80000000u) ? (k ^ 0x80000000u) : ~k;
  return __builtin_bit_cast(float, u);
}

// load a 16-element bf16 WMMA fragment: two b128 loads (K=base..+7, base+16..+23)
__device__ __forceinline__ v16bf ldfrag(const unsigned short* p) {
  uint4 lo = *(const uint4*)p;         // global_load_b128
  uint4 hi = *(const uint4*)(p + 16);  // global_load_b128
  v8u r;
  r[0] = lo.x; r[1] = lo.y; r[2] = lo.z; r[3] = lo.w;
  r[4] = hi.x; r[5] = hi.y; r[6] = hi.z; r[7] = hi.w;
  return __builtin_bit_cast(v16bf, r);
}

// ---------------- pre-conversion kernels ------------------------------------
// pack fp32 pairs -> bf16x2 dwords
__global__ void cvt_f32_bf16(const float* __restrict__ in,
                             unsigned* __restrict__ out, int npairs) {
  int i = blockIdx.x * 256 + threadIdx.x;
  if (i >= npairs) return;
  float2 v = ((const float2*)in)[i];
  out[i] = bf16bits(v.x) | (bf16bits(v.y) << 16);
}

// W[K][M] fp32 -> Wt[MP][K] bf16, zero-padded rows m in [M, MP)
__global__ void wt_cvt_bf16(const float* __restrict__ W,
                            unsigned short* __restrict__ Wt,
                            int K, int M, int MP) {
  int i = blockIdx.x * 256 + threadIdx.x;
  if (i >= MP * K) return;
  int m = i / K, k = i % K;
  float v = (m < M) ? W[(size_t)k * M + m] : 0.f;
  Wt[i] = (unsigned short)bf16bits(v);
}

// ---------------- WMMA bf16 GEMM: C[NxM] = A[NxK] @ Wt^T (+bias, relu) ------
// A: bf16 [N][K]; Wt: bf16 [MP][K] (pre-transposed, zero-padded).
// One wave computes a 32x16 output block (two row tiles share one B fragment).
// K is a template constant -> fully unrolled: 2*(K/32) v_wmma per wave.
template <int K>
__global__ void gemm_wmma_bf16(const unsigned short* __restrict__ A,
                               const unsigned short* __restrict__ Wt,
                               const float* __restrict__ bias,
                               float* __restrict__ C,
                               int M, int colTiles, int totalPairs,
                               int doRelu) {
  int p = blockIdx.x * 8 + (threadIdx.x >> 5);
  if (p >= totalPairs) return;                 // wave-uniform: EXEC all-ones
  int rowTile0 = (p / colTiles) * 2;
  int colTile  = p % colTiles;

  int lane = threadIdx.x & 31;
  int half = lane >> 4;                        // K-subrange select
  int rm   = lane & 15;                        // row (A) / col (B,C)

  const unsigned short* a0p = A  + (size_t)(rowTile0 * 16 + rm) * K + half * 8;
  const unsigned short* a1p = a0p + (size_t)16 * K;
  const unsigned short* bp  = Wt + (size_t)(colTile * 16 + rm) * K + half * 8;

  v8f acc0{}, acc1{};
#pragma unroll
  for (int kk = 0; kk < K; kk += 32) {
    v16bf a0 = ldfrag(a0p + kk);
    v16bf a1 = ldfrag(a1p + kk);
    v16bf b  = ldfrag(bp + kk);
    acc0 = __builtin_amdgcn_wmma_f32_16x16x32_bf16(
        false, a0, false, b, (short)0, acc0, false, false);
    acc1 = __builtin_amdgcn_wmma_f32_16x16x32_bf16(
        false, a1, false, b, (short)0, acc1, false, false);
  }

  int bcol = colTile * 16 + rm;
  if (bcol < M) {                              // only padded final layer masks
    float bv = bias ? bias[bcol] : 0.f;
#pragma unroll
    for (int i = 0; i < 8; ++i) {
      int r0 = rowTile0 * 16 + half * 8 + i;
      float v0 = acc0[i] + bv;
      float v1 = acc1[i] + bv;
      if (doRelu) { v0 = fmaxf(v0, 0.f); v1 = fmaxf(v1, 0.f); }
      C[(size_t)r0 * M + bcol]        = v0;
      C[(size_t)(r0 + 16) * M + bcol] = v1;
    }
  }
}

// ---------------- per-node attention logits: as[i]=h.a_s, ad[i]=h.a_d -------
__global__ void node_dots(const float* __restrict__ h,
                          const float* __restrict__ avs,
                          const float* __restrict__ avd,
                          float* __restrict__ outS, float* __restrict__ outD,
                          int D) {
  int node = blockIdx.x * 8 + (threadIdx.x >> 5);
  if (node >= NN) return;
  int lane = threadIdx.x & 31;
  const float* hp = h + (size_t)node * D;
  float s = 0.f, d = 0.f;
  for (int k = lane; k < D; k += 32) {
    float v = hp[k];
    s += v * avs[k];
    d += v * avd[k];
  }
#pragma unroll
  for (int off = 16; off > 0; off >>= 1) {
    s += __shfl_xor(s, off, 32);
    d += __shfl_xor(d, off, 32);
  }
  if (lane == 0) { outS[node] = s; outD[node] = d; }
}

// ---------------- edge pass 1: segment max of leaky-relu logits -------------
__global__ void edge_max(const int* __restrict__ src, const int* __restrict__ dst,
                         const float* __restrict__ as_, const float* __restrict__ ad_,
                         unsigned* __restrict__ mkey, int Etot) {
  int e = blockIdx.x * 256 + threadIdx.x;
  if (e >= Etot) return;
  int s, d;
  if (e < EE) { s = src[e]; d = dst[e]; } else { s = d = e - EE; }
  float a = as_[s] + ad_[d];
  a = (a > 0.f) ? a : 0.2f * a;
  atomicMax(&mkey[d], fkey(a));
}

// ---------------- edge pass 2: e=exp(a-m), segment sum ----------------------
__global__ void edge_expsum(const int* __restrict__ src, const int* __restrict__ dst,
                            const float* __restrict__ as_, const float* __restrict__ ad_,
                            const unsigned* __restrict__ mkey,
                            float* __restrict__ earr, float* __restrict__ ssum, int Etot) {
  int e = blockIdx.x * 256 + threadIdx.x;
  if (e >= Etot) return;
  int s, d;
  if (e < EE) { s = src[e]; d = dst[e]; } else { s = d = e - EE; }
  float a = as_[s] + ad_[d];
  a = (a > 0.f) ? a : 0.2f * a;
  float ex = __expf(a - funkey(mkey[d]));
  earr[e] = ex;
  atomicAdd(&ssum[d], ex);
}

// ---------------- edge aggregate: out[dst] += h[src]*coef (one wave/edge) ---
__global__ void edge_aggregate(const int* __restrict__ src, const int* __restrict__ dst,
                               const float* __restrict__ h,
                               const float* __restrict__ earr,
                               const float* __restrict__ ssum,
                               float* __restrict__ out, int D, int Etot) {
  int gid = blockIdx.x * 256 + threadIdx.x;
  int e = gid >> 5;
  if (e >= Etot) return;
  int lane = gid & 31;
  int s, d;
  if (e < EE) { s = src[e]; d = dst[e]; } else { s = d = e - EE; }
  float c = 1.0f;
  if (earr) c = earr[e] / (ssum[d] + 1e-16f);
  if (D == 128) {
    float4 v = ((const float4*)(h + (size_t)s * 128))[lane];   // global_load_b128
    float* o = out + (size_t)d * 128 + lane * 4;
    atomicAdd(o + 0, v.x * c);
    atomicAdd(o + 1, v.y * c);
    atomicAdd(o + 2, v.z * c);
    atomicAdd(o + 3, v.w * c);
  } else if (D == 64) {
    float2 v = ((const float2*)(h + (size_t)s * 64))[lane];
    float* o = out + (size_t)d * 64 + lane * 2;
    atomicAdd(o + 0, v.x * c);
    atomicAdd(o + 1, v.y * c);
  } else {
    for (int k = lane; k < D; k += 32)
      atomicAdd(&out[(size_t)d * D + k], h[(size_t)s * D + k] * c);
  }
}

// ---------------- bias + relu in place --------------------------------------
__global__ void bias_relu(float* __restrict__ h, const float* __restrict__ b, int D) {
  size_t i = (size_t)blockIdx.x * 256 + threadIdx.x;
  if (i >= (size_t)NN * D) return;
  float v = h[i] + b[i % D];
  h[i] = fmaxf(v, 0.f);
}

// ---------------- SAG score + tanh gate (one wave/node) ---------------------
__global__ void sag_gate(float* __restrict__ h, const float* __restrict__ agg,
                         const float* __restrict__ wpr, const float* __restrict__ wpo,
                         const float* __restrict__ bp) {
  int node = blockIdx.x * 8 + (threadIdx.x >> 5);
  if (node >= NN) return;
  int lane = threadIdx.x & 31;
  const float* ap = agg + (size_t)node * 128;
  float* hp = h + (size_t)node * 128;
  float s = 0.f;
  for (int k = lane; k < 128; k += 32)
    s += ap[k] * wpr[k] + hp[k] * wpo[k];
#pragma unroll
  for (int off = 16; off > 0; off >>= 1) s += __shfl_xor(s, off, 32);
  float t = tanhf(s + bp[0]);
  for (int k = lane; k < 128; k += 32) hp[k] *= t;
}

// ---------------- host-side orchestration -----------------------------------
// convert A (fp32 [NN][K]) and W (fp32 [K][M]) then run WMMA GEMM
static void launch_gemm(const float* A, const float* W, const float* bias, float* C,
                        int K, int M, int relu,
                        unsigned short* Abf, unsigned short* Wtbf,
                        hipStream_t stream) {
  int npairs = (NN * K) / 2;
  cvt_f32_bf16<<<(npairs + 255) / 256, 256, 0, stream>>>(A, (unsigned*)Abf, npairs);
  int MP = ((M + 15) / 16) * 16;
  wt_cvt_bf16<<<(MP * K + 255) / 256, 256, 0, stream>>>(W, Wtbf, K, M, MP);

  int colTiles = MP / 16;
  int totalPairs = ((NN / 16) / 2) * colTiles;   // 2500 row tiles -> 1250 pairs
  int blocks = (totalPairs + 7) / 8;
  if (K == 128)
    gemm_wmma_bf16<128><<<blocks, 256, 0, stream>>>(Abf, Wtbf, bias, C, M,
                                                    colTiles, totalPairs, relu);
  else
    gemm_wmma_bf16<64><<<blocks, 256, 0, stream>>>(Abf, Wtbf, bias, C, M,
                                                   colTiles, totalPairs, relu);
}

static void gat_layer(const float* hin, int K, int D,
                      const float* W, const float* avs, const float* avd,
                      const float* bias, const int* src, const int* dst,
                      float* lin, float* out, float* earr,
                      float* asn, float* adn, unsigned* mkey, float* ssum,
                      unsigned short* Abf, unsigned short* Wtbf,
                      hipStream_t stream) {
  const int Etot = EE + NN;
  launch_gemm(hin, W, nullptr, lin, K, D, 0, Abf, Wtbf, stream);
  node_dots<<<(NN + 7) / 8, 256, 0, stream>>>(lin, avs, avd, asn, adn, D);
  hipMemsetAsync(out,  0, (size_t)NN * D * sizeof(float), stream);
  hipMemsetAsync(mkey, 0, (size_t)NN * sizeof(unsigned), stream);
  hipMemsetAsync(ssum, 0, (size_t)NN * sizeof(float), stream);
  edge_max<<<(Etot + 255) / 256, 256, 0, stream>>>(src, dst, asn, adn, mkey, Etot);
  edge_expsum<<<(Etot + 255) / 256, 256, 0, stream>>>(src, dst, asn, adn, mkey,
                                                      earr, ssum, Etot);
  edge_aggregate<<<((Etot * 32) + 255) / 256, 256, 0, stream>>>(src, dst, lin, earr,
                                                                ssum, out, D, Etot);
  bias_relu<<<(int)(((size_t)NN * D + 255) / 256), 256, 0, stream>>>(out, bias, D);
}

extern "C" void kernel_launch(void* const* d_in, const int* in_sizes, int n_in,
                              void* d_out, int out_size, void* d_ws, size_t ws_size,
                              hipStream_t stream) {
  const float* x   = (const float*)d_in[0];
  const int*   ei  = (const int*)d_in[1];
  const int*   src = ei;
  const int*   dst = ei + EE;
  const float* W1  = (const float*)d_in[4];
  const float* a1s = (const float*)d_in[5];
  const float* a1d = (const float*)d_in[6];
  const float* b1  = (const float*)d_in[7];
  const float* W2  = (const float*)d_in[8];
  const float* a2s = (const float*)d_in[9];
  const float* a2d = (const float*)d_in[10];
  const float* b2  = (const float*)d_in[11];
  const float* W3  = (const float*)d_in[12];
  const float* a3s = (const float*)d_in[13];
  const float* a3d = (const float*)d_in[14];
  const float* b3  = (const float*)d_in[15];
  const float* wpr = (const float*)d_in[16];
  const float* wpo = (const float*)d_in[17];
  const float* bp  = (const float*)d_in[18];
  const float* Wl  = (const float*)d_in[19];
  const float* bl  = (const float*)d_in[20];

  // workspace layout (256B aligned)
  char* ws = (char*)d_ws;
  size_t off = 0;
  auto alloc = [&](size_t bytes) {
    char* p = ws + off;
    off += (bytes + 255) & ~(size_t)255;
    return p;
  };
  float*    B0   = (float*)alloc((size_t)NN * 128 * sizeof(float));
  float*    B1   = (float*)alloc((size_t)NN * 128 * sizeof(float));
  float*    B2   = (float*)alloc((size_t)NN * 128 * sizeof(float));
  float*    earr = (float*)alloc((size_t)(EE + NN) * sizeof(float));
  float*    asn  = (float*)alloc((size_t)NN * sizeof(float));
  float*    adn  = (float*)alloc((size_t)NN * sizeof(float));
  unsigned* mkey = (unsigned*)alloc((size_t)NN * sizeof(unsigned));
  float*    ssum = (float*)alloc((size_t)NN * sizeof(float));
  unsigned short* Abf  = (unsigned short*)alloc((size_t)NN * 128 * sizeof(unsigned short));
  unsigned short* Wtbf = (unsigned short*)alloc((size_t)128 * 128 * sizeof(unsigned short));

  // layer 1: 128 -> 64   (in=x, lin=B0, out=B1)
  gat_layer(x, 128, 64, W1, a1s, a1d, b1, src, dst, B0, B1, earr, asn, adn, mkey,
            ssum, Abf, Wtbf, stream);
  // layer 2: 64 -> 128   (in=B1, lin=B2, out=B0)
  gat_layer(B1, 64, 128, W2, a2s, a2d, b2, src, dst, B2, B0, earr, asn, adn, mkey,
            ssum, Abf, Wtbf, stream);
  // layer 3: 128 -> 128  (in=B0, lin=B1, out=B2)
  gat_layer(B0, 128, 128, W3, a3s, a3d, b3, src, dst, B1, B2, earr, asn, adn, mkey,
            ssum, Abf, Wtbf, stream);

  // SAGPooling score: agg over raw edges (no self-loops, coef = 1) into B1
  hipMemsetAsync(B1, 0, (size_t)NN * 128 * sizeof(float), stream);
  edge_aggregate<<<((EE * 32) + 255) / 256, 256, 0, stream>>>(
      src, dst, B2, nullptr, nullptr, B1, 128, EE);
  sag_gate<<<(NN + 7) / 8, 256, 0, stream>>>(B2, B1, wpr, wpo, bp);

  // final linear 128 -> 10 with relu
  launch_gemm(B2, Wl, bl, (float*)d_out, 128, DOUT_FINAL, 1, Abf, Wtbf, stream);
}
